// MultiHeadedAttention_60129542144350
// MI455X (gfx1250) — matrix operand order repaired and checked by reading, not compiled, not run on hardware
//
#include <hip/hip_runtime.h>
#include <math.h>

// Problem constants (match reference)
#define HEADS 4
#define DKV   64      // dk == dv == 64
#define DMODEL 256
#define BATCH 8
#define SEQ   2048
#define LN_EPS 1e-5f
#define NEG_INF (-__builtin_inff())

typedef __bf16 bf16;
typedef __attribute__((ext_vector_type(16))) __bf16 v16bf;
typedef __attribute__((ext_vector_type(8)))  __bf16 v8bf;
typedef __attribute__((ext_vector_type(8)))  float  v8f;
typedef __attribute__((ext_vector_type(4)))  int    v4i;

// ---- CDNA5 async load-to-LDS support (guarded; falls back to sync loads) ----
#if defined(__has_builtin)
#  if __has_builtin(__builtin_amdgcn_global_load_async_to_lds_b128)
#    define HAVE_ASYNC_LDS 1
#  endif
#  if __has_builtin(__builtin_amdgcn_s_wait_asynccnt)
#    define HAVE_WAIT_ASYNC 1
#  endif
#endif
#ifndef HAVE_ASYNC_LDS
#  define HAVE_ASYNC_LDS 0
#endif
#ifndef HAVE_WAIT_ASYNC
#  define HAVE_WAIT_ASYNC 0
#endif

#if HAVE_WAIT_ASYNC
#  define WAIT_ASYNC(n) __builtin_amdgcn_s_wait_asynccnt(n)
#else
#  define WAIT_ASYNC(n) asm volatile("s_wait_asynccnt " #n ::: "memory")
#endif

#if HAVE_ASYNC_LDS
// one GLOBAL_LOAD_ASYNC_TO_LDS_B128 (16 bytes) per lane
// prototype (from hipcc diagnostic): (v4i*, v4i*, imm offset, imm cpol)
__device__ __forceinline__ void async_copy16(const void* g, void* l) {
  __builtin_amdgcn_global_load_async_to_lds_b128((v4i*)g, (v4i*)l, 0, 0);
}
// stage one 32-byte per-lane chunk (two b128s) -> ASYNCcnt += 2 per wave
__device__ __forceinline__ void stage32(const bf16* src, bf16* dst) {
  async_copy16(src, dst);
  async_copy16(src + 8, dst + 8);
}
#endif

__device__ __forceinline__ v8f wmma_bf16(v16bf a, v16bf b, v8f c) {
  // emits v_wmma_f32_16x16x32_bf16
  return __builtin_amdgcn_wmma_f32_16x16x32_bf16(false, a, false, b, (short)0, c, false, false);
}

// Build A-fragment (16x32, 16-bit layout) from an fp32 row:
// lane<16: K = kbase+0..15 ; lane>=16: K = kbase+16..31
__device__ __forceinline__ v16bf a_frag_f32(const float* row, int kbase, int lane) {
  int o = kbase + ((lane & 16) ? 16 : 0);
  v16bf r;
#pragma unroll
  for (int i = 0; i < 8; ++i) {
    r[i]     = (bf16)row[o + i];
    r[8 + i] = (bf16)row[o + 8 + i];
  }
  return r;
}

__device__ __forceinline__ v16bf a_frag_bf16(const bf16* row, int kbase, int lane) {
  int o = kbase + ((lane & 16) ? 16 : 0);
  v8bf c0 = *(const v8bf*)(row + o);
  v8bf c1 = *(const v8bf*)(row + o + 8);
  v16bf r;
#pragma unroll
  for (int i = 0; i < 8; ++i) { r[i] = c0[i]; r[8 + i] = c1[i]; }
  return r;
}

// ---------------- Kernel 0: weight transpose + bf16 convert ----------------
// w: [256(k), 256(n)] fp32 row-major  ->  wt: [256(n), 256(k)] bf16
__global__ __launch_bounds__(256) void wt_kernel(const float* __restrict__ w,
                                                 bf16* __restrict__ wt) {
  int idx = blockIdx.x * 256 + threadIdx.x;   // 65536 total
  int n = idx >> 8, k = idx & 255;
  wt[n * DMODEL + k] = (bf16)w[k * DMODEL + n];
}

// ---------------- Kernel 1: QKV projection GEMM (M=16384,N=256,K=256) ------
// 4 waves / block, each wave owns 64 output columns (== one head).
__global__ __launch_bounds__(128) void qkv_kernel(const float* __restrict__ x,
                                                  const bf16* __restrict__ wt,
                                                  const float* __restrict__ bias,
                                                  bf16* __restrict__ dst_hld,  // [B,H,L,DKV] (q/k)
                                                  bf16* __restrict__ dst_vT,   // [B,H,DKV,L] (v)
                                                  int is_v) {
  int tid = threadIdx.x, wave = tid >> 5, lane = tid & 31;
  int row0 = blockIdx.x * 16;
  int b = row0 >> 11, l0 = row0 & (SEQ - 1);
  const float* arow = x + (size_t)(row0 + (lane & 15)) * DMODEL;
  int ncol0 = wave * 64;

  v8f acc[4] = {};
#pragma unroll
  for (int kk = 0; kk < 8; ++kk) {
    int kbase = kk * 32;
    v16bf a = a_frag_f32(arow, kbase, lane);
#pragma unroll
    for (int nt = 0; nt < 4; ++nt) {
      int ncol = ncol0 + nt * 16 + (lane & 15);
      v16bf bfrag = *(const v16bf*)(wt + (size_t)ncol * DMODEL + kbase + ((lane & 16) ? 16 : 0));
      acc[nt] = wmma_bf16(a, bfrag, acc[nt]);
    }
  }

  int hi = (lane & 16) ? 8 : 0;
#pragma unroll
  for (int nt = 0; nt < 4; ++nt) {
    int ncol = ncol0 + nt * 16 + (lane & 15);
    float bb = bias[ncol];
    int h = ncol >> 6, d = ncol & 63;
    if (!is_v) {
      bf16* dst = dst_hld + (((size_t)(b * HEADS + h) * SEQ + l0) * DKV) + d;
#pragma unroll
      for (int r = 0; r < 8; ++r)
        dst[(size_t)(r + hi) * DKV] = (bf16)(acc[nt][r] + bb);
    } else {
      v8bf pack;
#pragma unroll
      for (int r = 0; r < 8; ++r) pack[r] = (bf16)(acc[nt][r] + bb);
      *(v8bf*)(dst_vT + ((size_t)(b * HEADS + h) * DKV + d) * SEQ + l0 + hi) = pack;
    }
  }
}

// ---------------- Kernel 2: fused scores + masked softmax + attn@V ---------
// One workgroup = 16 query rows for one (b,h). 8 waves.
// Phase1: S[16][2048] logits in LDS (each wave owns 256 keys).
// Phase2: row softmax in LDS, write probs (fp32) to d_out attn region.
// Phase3: O = P @ V with bf16 WMMA; V tiles double-buffered into LDS with
//         GLOBAL_LOAD_ASYNC_TO_LDS_B128 + s_wait_asynccnt (ASYNCcnt path).
__global__ __launch_bounds__(256) void attn_kernel(const bf16* __restrict__ qh,
                                                   const bf16* __restrict__ kh,
                                                   const bf16* __restrict__ vhT,
                                                   const unsigned char* __restrict__ mask,
                                                   float* __restrict__ attn_out,
                                                   bf16* __restrict__ attnO) {
  __shared__ float S[16][SEQ];        // 128 KB (CDNA5 WGP LDS = 320 KB)
  __shared__ float red[16][17];
  __shared__ float rowstat[16];
  __shared__ float comb[4][256];
  __shared__ __align__(32) bf16 vstage[2][8][512];  // 16 KB: [buf][wave][16 dv x 32 k]

  int tid = threadIdx.x, wave = tid >> 5, lane = tid & 31;
  int wg = blockIdx.x;
  int qt = wg & 127;            // 128 q-tiles per (b,h)
  int bh = wg >> 7;
  int b = bh >> 2, h = bh & 3;

  const bf16* qrowp = qh + ((size_t)(b * HEADS + h) * SEQ + qt * 16 + (lane & 15)) * DKV;
  v16bf aq0 = a_frag_bf16(qrowp, 0, lane);
  v16bf aq1 = a_frag_bf16(qrowp, 32, lane);

  const bf16* khb = kh + (size_t)(b * HEADS + h) * SEQ * DKV;
  int hi = (lane & 16) ? 8 : 0;
  int bo = (lane & 16) ? 16 : 0;

  // ---- Phase 1: scores ----
  int kw0 = wave * 256;
  for (int nt = 0; nt < 16; ++nt) {
    int key = kw0 + nt * 16 + (lane & 15);
    const bf16* krow = khb + (size_t)key * DKV;
    __builtin_prefetch(krow + 16 * DKV, 0, 0);   // global_prefetch_b8, next tile
    v16bf b0 = *(const v16bf*)(krow + bo);
    v16bf b1 = *(const v16bf*)(krow + 32 + bo);
    v8f acc = {};
    acc = wmma_bf16(aq0, b0, acc);
    acc = wmma_bf16(aq1, b1, acc);
    bool msk = mask[b * SEQ + key] != 0;
#pragma unroll
    for (int r = 0; r < 8; ++r)
      S[r + hi][key] = msk ? NEG_INF : acc[r] * 0.125f;   // 1/sqrt(64)
  }
  __syncthreads();

  // ---- Phase 2: softmax over each of the 16 rows ----
  {
    int r = tid >> 4, c0 = tid & 15;
    float m = NEG_INF;
    for (int j = 0; j < 128; ++j) m = fmaxf(m, S[r][c0 + (j << 4)]);
    red[r][c0] = m;
    __syncthreads();
    float rowmax = NEG_INF;
#pragma unroll
    for (int i = 0; i < 16; ++i) rowmax = fmaxf(rowmax, red[r][i]);
    __syncthreads();
    float sum = 0.f;
    for (int j = 0; j < 128; ++j) {
      int c = c0 + (j << 4);
      float e = __expf(S[r][c] - rowmax);
      S[r][c] = e;
      sum += e;
    }
    red[r][c0] = sum;
    __syncthreads();
    float rowsum = 0.f;
#pragma unroll
    for (int i = 0; i < 16; ++i) rowsum += red[r][i];
    if (c0 == 0) rowstat[r] = 1.0f / rowsum;
  }
  __syncthreads();

  // normalize in LDS + stream probabilities to global (coalesced)
  {
    float* aout = attn_out + ((size_t)(h * BATCH + b) * SEQ + qt * 16) * SEQ;
    for (int j = 0; j < 128; ++j) {
      int idx = tid + (j << 8);
      int rr = idx >> 11, cc = idx & (SEQ - 1);
      float p = S[rr][cc] * rowstat[rr];
      S[rr][cc] = p;
      aout[(size_t)rr * SEQ + cc] = p;
    }
  }
  __syncthreads();

  // ---- Phase 3: O[16x64] = P[16x2048] @ V[2048x64] ----
  int nt = wave & 3, khalf = wave >> 2;
  int dv0 = nt * 16;
  int dv = dv0 + (lane & 15);
  const bf16* vbase = vhT + ((size_t)(b * HEADS + h) * DKV) * SEQ;   // [dv][SEQ]
  int qrow = lane & 15;
  int alo = (lane & 16) ? 16 : 0;
  v8f acc = {};

#if HAVE_ASYNC_LDS
  // async double-buffered staging of the 16dv x 32k bf16 V tile (1 KB / wave)
  int srow = lane >> 1;                 // 0..15  (local dv)
  int scol = (lane & 1) * 16;           // 0 or 16 (k elements)
  const bf16* vsrc = vbase + (size_t)(dv0 + srow) * SEQ + khalf * 1024 + scol;
  bf16* vdst0 = &vstage[0][wave][srow * 32 + scol];
  bf16* vdst1 = &vstage[1][wave][srow * 32 + scol];
  stage32(vsrc, vdst0);                 // preload ks = 0
  for (int ks = 0; ks < 32; ++ks) {
    if (ks + 1 < 32) {
      stage32(vsrc + (ks + 1) * 32, (ks & 1) ? vdst0 : vdst1);
      WAIT_ASYNC(2);                    // current tile complete; next in flight
    } else {
      WAIT_ASYNC(0);
    }
    asm volatile("" ::: "memory");      // keep LDS reads below the wait
    int kbase = khalf * 1024 + ks * 32;
    v16bf a;
#pragma unroll
    for (int i = 0; i < 8; ++i) {
      a[i]     = (bf16)S[qrow][kbase + alo + i];
      a[8 + i] = (bf16)S[qrow][kbase + alo + 8 + i];
    }
    v16bf bfr = *(const v16bf*)&vstage[ks & 1][wave][(lane & 15) * 32 + bo];
    acc = wmma_bf16(a, bfr, acc);
  }
#else
  const bf16* vrow = vbase + (size_t)dv * SEQ;
  for (int ks = 0; ks < 32; ++ks) {
    int kbase = khalf * 1024 + ks * 32;
    v16bf a;
#pragma unroll
    for (int i = 0; i < 8; ++i) {
      a[i]     = (bf16)S[qrow][kbase + alo + i];
      a[8 + i] = (bf16)S[qrow][kbase + alo + 8 + i];
    }
    v16bf bfr = *(const v16bf*)(vrow + kbase + bo);
    acc = wmma_bf16(a, bfr, acc);
  }
#endif

  if (khalf == 1) {
#pragma unroll
    for (int r = 0; r < 8; ++r) comb[nt][r * 32 + lane] = acc[r];
  }
  __syncthreads();
  if (khalf == 0) {
    bf16* orow = attnO + ((size_t)b * SEQ + qt * 16) * (HEADS * DKV) + h * DKV + dv;
#pragma unroll
    for (int r = 0; r < 8; ++r) {
      float v = acc[r] + comb[nt][r * 32 + lane];
      orow[(size_t)(r + hi) * (HEADS * DKV)] = (bf16)v;
    }
  }
}

// ---------------- Kernel 3: out proj + residual + LayerNorm ----------------
__global__ __launch_bounds__(128) void out_kernel(const bf16* __restrict__ attnO,
                                                  const bf16* __restrict__ wfT,
                                                  const float* __restrict__ bf_,
                                                  const float* __restrict__ resid,
                                                  const float* __restrict__ gamma,
                                                  const float* __restrict__ beta,
                                                  float* __restrict__ out) {
  __shared__ float xs[16][DMODEL];    // 16 KB
  __shared__ float rsum[16][8];
  __shared__ float rsq[16][8];
  __shared__ float mu[16];
  __shared__ float var_[16];

  int tid = threadIdx.x, wave = tid >> 5, lane = tid & 31;
  int row0 = blockIdx.x * 16;
  const bf16* arow = attnO + (size_t)(row0 + (lane & 15)) * DMODEL;
  int ncol0 = wave * 64;

  v8f acc[4] = {};
#pragma unroll
  for (int kk = 0; kk < 8; ++kk) {
    int kbase = kk * 32;
    v16bf a = a_frag_bf16(arow, kbase, lane);
#pragma unroll
    for (int nt = 0; nt < 4; ++nt) {
      int ncol = ncol0 + nt * 16 + (lane & 15);
      v16bf bfr = *(const v16bf*)(wfT + (size_t)ncol * DMODEL + kbase + ((lane & 16) ? 16 : 0));
      acc[nt] = wmma_bf16(a, bfr, acc[nt]);
    }
  }

  int hi = (lane & 16) ? 8 : 0;
#pragma unroll
  for (int nt = 0; nt < 4; ++nt) {
    int ncol = ncol0 + nt * 16 + (lane & 15);
    float bb = bf_[ncol];
#pragma unroll
    for (int r = 0; r < 8; ++r) {
      int m = r + hi;
      xs[m][ncol] = acc[nt][r] + bb + resid[(size_t)(row0 + m) * DMODEL + ncol];
    }
  }
  __syncthreads();

  int rr = tid >> 3, cc = tid & 7;
  float s = 0.f, sq = 0.f;
  for (int j = 0; j < 32; ++j) {
    float v = xs[rr][cc + (j << 3)];
    s += v; sq += v * v;
  }
  rsum[rr][cc] = s; rsq[rr][cc] = sq;
  __syncthreads();
  if (cc == 0) {
    float ts = 0.f, tq = 0.f;
#pragma unroll
    for (int i = 0; i < 8; ++i) { ts += rsum[rr][i]; tq += rsq[rr][i]; }
    float m_ = ts / DMODEL;
    mu[rr] = m_;
    var_[rr] = tq / DMODEL - m_ * m_;
  }
  __syncthreads();

  for (int j = 0; j < 32; ++j) {
    int idx = tid + (j << 7);
    int m = idx >> 8, n = idx & 255;
    float v = (xs[m][n] - mu[m]) * rsqrtf(var_[m] + LN_EPS);
    out[(size_t)(row0 + m) * DMODEL + n] = gamma[n] * v + beta[n];
  }
}

// ---------------- Host launcher ----------------
extern "C" void kernel_launch(void* const* d_in, const int* in_sizes, int n_in,
                              void* d_out, int out_size, void* d_ws, size_t ws_size,
                              hipStream_t stream) {
  (void)in_sizes; (void)n_in; (void)out_size; (void)ws_size;
  const float* q    = (const float*)d_in[0];
  const float* k    = (const float*)d_in[1];
  const float* v    = (const float*)d_in[2];
  const unsigned char* mask = (const unsigned char*)d_in[3];  // numpy bool = 1 byte
  const float* wq   = (const float*)d_in[4];
  const float* bq   = (const float*)d_in[5];
  const float* wk   = (const float*)d_in[6];
  const float* bk   = (const float*)d_in[7];
  const float* wv   = (const float*)d_in[8];
  const float* bv   = (const float*)d_in[9];
  const float* wf   = (const float*)d_in[10];
  const float* bf_  = (const float*)d_in[11];
  const float* gamma = (const float*)d_in[12];
  const float* beta  = (const float*)d_in[13];

  float* out      = (float*)d_out;
  float* attn_out = out + (size_t)BATCH * SEQ * DMODEL;

  // Workspace carving (bf16): 4 weight transposes + qh/kh/vhT/attnO (~34 MB)
  bf16* wqT = (bf16*)d_ws;
  bf16* wkT = wqT + 65536;
  bf16* wvT = wkT + 65536;
  bf16* wfT = wvT + 65536;
  bf16* qh  = wfT + 65536;
  const size_t HLD = (size_t)BATCH * HEADS * SEQ * DKV;   // 4,194,304
  bf16* kh   = qh  + HLD;
  bf16* vhT  = kh  + HLD;
  bf16* attnO = vhT + HLD;

  wt_kernel<<<256, 256, 0, stream>>>(wq, wqT);
  wt_kernel<<<256, 256, 0, stream>>>(wk, wkT);
  wt_kernel<<<256, 256, 0, stream>>>(wv, wvT);
  wt_kernel<<<256, 256, 0, stream>>>(wf, wfT);

  qkv_kernel<<<1024, 128, 0, stream>>>(q, wqT, bq, qh,  nullptr, 0);
  qkv_kernel<<<1024, 128, 0, stream>>>(k, wkT, bk, kh,  nullptr, 0);
  qkv_kernel<<<1024, 128, 0, stream>>>(v, wvT, bv, nullptr, vhT, 1);

  attn_kernel<<<BATCH * HEADS * (SEQ / 16), 256, 0, stream>>>(qh, kh, vhT, mask, attn_out, attnO);

  out_kernel<<<1024, 128, 0, stream>>>(attnO, wfT, bf_, q, gamma, beta, out);
}
